// PointTransformer_FPADV_75642964017686
// MI455X (gfx1250) — matrix operand order repaired and checked by reading, not compiled, not run on hardware
//
#include <hip/hip_runtime.h>

// ============================================================================
// PCT (Point Cloud Transformer) forward for MI455X / gfx1250.
// All GEMM-like work (conv1d, local_op, attention, fuse, FP convs, logits)
// runs through one strided batched GEMM kernel using V_WMMA_F32_16X16X32_BF16
// (f32 -> bf16 LDS staging, f32 accumulate). Wave32, 4 waves / 128-thread
// block, 32x32 block tile, K-step 64 (2 WMMA per barrier pair). Staging tier
// selection is block-uniform -> scalar branches, no exec divergence on
// interior tiles. Workspace requirement: ~400 MB.
// ============================================================================

typedef __attribute__((ext_vector_type(16))) __bf16 v16bf;
typedef __attribute__((ext_vector_type(8)))  float  v8f;

union bfpack { __bf16 h[2]; unsigned u; };

// ---------------------------------------------------------------------------
// Generic strided batched GEMM:
//   C[m][n] = sum_k A[m*sAm + k*sAk] * B[n*sBn + k*sBk]   (+ bias[m])
// stored at C[m*sCm + n*sCn]; per-batch offsets bA/bB/bC (0 => shared).
// ---------------------------------------------------------------------------
__global__ void k_gemm_bf16(
    float* __restrict__ C, const float* __restrict__ A, const float* __restrict__ Bm,
    const float* __restrict__ bias,
    int M, int N, int K,
    long sAm, long sAk, long bA,
    long sBn, long sBk, long bB,
    long sCm, long sCn, long bC)
{
  __shared__ __bf16 As[32][72];   // 64 k + 8 pad (144B row, 4B aligned pairs)
  __shared__ __bf16 Bs[32][72];

  const int tid  = threadIdx.x;
  const int lane = tid & 31;
  const int wave = tid >> 5;
  const int wm   = (wave >> 1) * 16;       // wave tile inside 32x32 block tile
  const int wn   = (wave &  1) * 16;
  const int bm   = blockIdx.y * 32;
  const int bn   = blockIdx.x * 32;
  const float* Ab = A  + (long)blockIdx.z * bA;
  const float* Bb = Bm + (long)blockIdx.z * bB;
  float*       Cb = C  + (long)blockIdx.z * bC;

  const int row = tid >> 2;                // 0..31 staging row
  const int kc  = (tid & 3) * 16;          // 0,16,32,48

  const int  am    = bm + row;
  const bool amok  = am < M;
  const long abase = (long)(amok ? am : 0) * sAm;
  const int  bnr   = bn + row;
  const bool bnok  = bnr < N;
  const long bbase = (long)(bnok ? bnr : 0) * sBn;

  // block-uniform tier eligibility
  const bool fullM = (bm + 32 <= M);
  const bool fullN = (bn + 32 <= N);
  const bool vecA  = (sAk == 1) && ((sAm & 3) == 0) && fullM;
  const bool vecB  = (sBk == 1) && ((sBn & 3) == 0) && fullN;

  v8f acc = {};

  for (int k0 = 0; k0 < K; k0 += 64) {
    const bool fullK = (k0 + 64 <= K);       // uniform per iteration
    // ---------------- stage A (32 m-rows x 64 k) ----------------
    if (vecA && fullK) {
      // tier 1: contiguous, aligned, in-bounds -> global_load_b128
      const float4* p = (const float4*)(Ab + abase + k0 + kc);
#pragma unroll
      for (int q = 0; q < 4; ++q) {
        const float4 v = p[q];
        bfpack p0, p1;
        p0.h[0] = (__bf16)v.x; p0.h[1] = (__bf16)v.y;
        p1.h[0] = (__bf16)v.z; p1.h[1] = (__bf16)v.w;
        unsigned* dst = (unsigned*)&As[row][kc + q * 4];
        dst[0] = p0.u; dst[1] = p1.u;
      }
    } else if (fullM && fullK) {
      // tier 2: strided but fully in-bounds -> unconditional batched loads
#pragma unroll
      for (int i = 0; i < 16; i += 2) {
        const float v1 = Ab[abase + (long)(k0 + kc + i)     * sAk];
        const float v2 = Ab[abase + (long)(k0 + kc + i + 1) * sAk];
        bfpack pk;
        pk.h[0] = (__bf16)v1; pk.h[1] = (__bf16)v2;
        *(unsigned*)&As[row][kc + i] = pk.u;
      }
    } else {
      // tier 3: edge -> clamped indices + value masking
#pragma unroll
      for (int i = 0; i < 16; i += 2) {
        const int k1 = k0 + kc + i, k2 = k1 + 1;
        float v1 = Ab[abase + (long)(k1 < K ? k1 : 0) * sAk];
        float v2 = Ab[abase + (long)(k2 < K ? k2 : 0) * sAk];
        if (!(amok && k1 < K)) v1 = 0.f;
        if (!(amok && k2 < K)) v2 = 0.f;
        bfpack pk;
        pk.h[0] = (__bf16)v1; pk.h[1] = (__bf16)v2;
        *(unsigned*)&As[row][kc + i] = pk.u;
      }
    }
    // ---------------- stage B (32 n-rows x 64 k) ----------------
    if (vecB && fullK) {
      const float4* p = (const float4*)(Bb + bbase + k0 + kc);
#pragma unroll
      for (int q = 0; q < 4; ++q) {
        const float4 v = p[q];
        bfpack p0, p1;
        p0.h[0] = (__bf16)v.x; p0.h[1] = (__bf16)v.y;
        p1.h[0] = (__bf16)v.z; p1.h[1] = (__bf16)v.w;
        unsigned* dst = (unsigned*)&Bs[row][kc + q * 4];
        dst[0] = p0.u; dst[1] = p1.u;
      }
    } else if (fullN && fullK) {
#pragma unroll
      for (int i = 0; i < 16; i += 2) {
        const float v1 = Bb[bbase + (long)(k0 + kc + i)     * sBk];
        const float v2 = Bb[bbase + (long)(k0 + kc + i + 1) * sBk];
        bfpack pk;
        pk.h[0] = (__bf16)v1; pk.h[1] = (__bf16)v2;
        *(unsigned*)&Bs[row][kc + i] = pk.u;
      }
    } else {
#pragma unroll
      for (int i = 0; i < 16; i += 2) {
        const int k1 = k0 + kc + i, k2 = k1 + 1;
        float v1 = Bb[bbase + (long)(k1 < K ? k1 : 0) * sBk];
        float v2 = Bb[bbase + (long)(k2 < K ? k2 : 0) * sBk];
        if (!(bnok && k1 < K)) v1 = 0.f;
        if (!(bnok && k2 < K)) v2 = 0.f;
        bfpack pk;
        pk.h[0] = (__bf16)v1; pk.h[1] = (__bf16)v2;
        *(unsigned*)&Bs[row][kc + i] = pk.u;
      }
    }
    // prefetch next K tile (global_prefetch_b8)
    if (k0 + 64 + kc < K) {
      __builtin_prefetch(Ab + abase + (long)(k0 + 64 + kc) * sAk, 0, 1);
      __builtin_prefetch(Bb + bbase + (long)(k0 + 64 + kc) * sBk, 0, 1);
    }
    __syncthreads();

    // ---- two WMMAs over the 64-wide LDS tile; contiguous k order per lane half (A and B alike) ----
#pragma unroll
    for (int h = 0; h < 2; ++h) {
      v16bf af, bfr;
      const int ml  = lane & 15;
      const int kbA = ((lane < 16) ? 0 : 16) + h * 32;  // A: hi lanes hold K=16..31
      const int kbB = ((lane < 16) ? 0 : 16) + h * 32;  // B: hi lanes hold K=16..31
#pragma unroll
      for (int j = 0; j < 16; ++j) {
        af[j]  = As[wm + ml][kbA + j];
        bfr[j] = Bs[wn + ml][kbB + j];
      }
      acc = __builtin_amdgcn_wmma_f32_16x16x32_bf16(
          false, af, false, bfr, (short)0, acc, false, false);
    }
    __syncthreads();
  }

  // ---- store: VGPR r, lanes 0-15 -> M=r, lanes 16-31 -> M=8+r ----
  const int nl  = lane & 15;
  const int mhi = (lane >> 4) * 8;
#pragma unroll
  for (int r = 0; r < 8; ++r) {
    const int m = bm + wm + mhi + r;
    const int n = bn + wn + nl;
    if (m < M && n < N) {
      float v = acc[r];
      if (bias) v += bias[m];
      Cb[(long)m * sCm + (long)n * sCn] = v;
    }
  }
}

// ---------------------------------------------------------------------------
// BatchNorm statistics: one block per channel c; elements at
// X[b*sb + c*sc + n], b in [0,Bc), n in [0,Nc). stats[c]=mean, stats[C+c]=rstd.
// ---------------------------------------------------------------------------
__global__ void k_bn_stats(const float* __restrict__ X, float* __restrict__ stats,
                           int C, int Bc, int Nc, long sc, long sb, float eps)
{
  const int c = blockIdx.x;
  __shared__ float ssum[256];
  __shared__ float ssq[256];
  float s = 0.f, q = 0.f;
  const long total = (long)Bc * Nc;
  for (long i = threadIdx.x; i < total; i += blockDim.x) {
    const long b = i / Nc, n = i % Nc;
    const float v = X[b * sb + (long)c * sc + n];
    s += v; q += v * v;
  }
  ssum[threadIdx.x] = s; ssq[threadIdx.x] = q;
  __syncthreads();
  for (int st = 128; st > 0; st >>= 1) {
    if ((int)threadIdx.x < st) {
      ssum[threadIdx.x] += ssum[threadIdx.x + st];
      ssq[threadIdx.x]  += ssq[threadIdx.x + st];
    }
    __syncthreads();
  }
  if (threadIdx.x == 0) {
    const float mean = ssum[0] / (float)total;
    const float var  = ssq[0] / (float)total - mean * mean;
    stats[c]     = mean;
    stats[C + c] = rsqrtf(var + eps);
  }
}

// y = [res +] act(g[c]*(x-mean[c])*rstd[c] + b[c]);  act 0=none 1=relu 2=leaky0.2
__global__ void k_bn_apply(const float* __restrict__ X, float* __restrict__ Y,
                           const float* __restrict__ res,
                           const float* __restrict__ stats,
                           const float* __restrict__ g, const float* __restrict__ bb,
                           int C, int Bc, int Nc,
                           long sc, long sb, long scO, long sbO, long scR, long sbR,
                           int act)
{
  const long total = (long)C * Bc * Nc;
  const long idx = (long)blockIdx.x * blockDim.x + threadIdx.x;
  if (idx >= total) return;
  const int  n = (int)(idx % Nc);
  const int  c = (int)((idx / Nc) % C);
  const long b = idx / ((long)Nc * C);
  float v = X[b * sb + (long)c * sc + n];
  v = g[c] * (v - stats[c]) * stats[C + c] + bb[c];
  if (act == 1)      v = v > 0.f ? v : 0.f;
  else if (act == 2) v = v > 0.f ? v : 0.2f * v;
  if (res) v += res[b * sbR + (long)c * scR + n];
  Y[b * sbO + (long)c * scO + n] = v;
}

// ---------------------------------------------------------------------------
// Farthest point sampling: one block per batch; sequential scan, argmax with
// first-occurrence tie-break. xyz [B,N,3], idx [B,S].
// ---------------------------------------------------------------------------
__global__ void k_fps(const float* __restrict__ xyz, int* __restrict__ idx,
                      int N, int S)
{
  __shared__ float dist[2048];
  __shared__ float bval[256];
  __shared__ int   bidx[256];
  __shared__ int   cur;
  const int b = blockIdx.x;
  const float* P = xyz + (long)b * N * 3;
  for (int i = threadIdx.x; i < N; i += blockDim.x) dist[i] = 1e10f;
  if (threadIdx.x == 0) cur = 0;
  __syncthreads();
  for (int s = 0; s < S; ++s) {
    const int c = cur;
    if (threadIdx.x == 0) idx[(long)b * S + s] = c;
    const float cx = P[c * 3 + 0], cy = P[c * 3 + 1], cz = P[c * 3 + 2];
    float best = -1.f; int bi = N;
    for (int i = threadIdx.x; i < N; i += blockDim.x) {
      const float dx = P[i * 3 + 0] - cx;
      const float dy = P[i * 3 + 1] - cy;
      const float dz = P[i * 3 + 2] - cz;
      const float dd = fminf(dist[i], dx * dx + dy * dy + dz * dz);
      dist[i] = dd;
      if (dd > best) { best = dd; bi = i; }
    }
    bval[threadIdx.x] = best; bidx[threadIdx.x] = bi;
    __syncthreads();
    for (int st = 128; st > 0; st >>= 1) {
      if ((int)threadIdx.x < st) {
        const float v2 = bval[threadIdx.x + st];
        const int   i2 = bidx[threadIdx.x + st];
        if (v2 > bval[threadIdx.x] ||
            (v2 == bval[threadIdx.x] && i2 < bidx[threadIdx.x])) {
          bval[threadIdx.x] = v2; bidx[threadIdx.x] = i2;
        }
      }
      __syncthreads();
    }
    if (threadIdx.x == 0) cur = bidx[0];
    __syncthreads();
  }
}

__global__ void k_gather3(const float* __restrict__ xyz, const int* __restrict__ fi,
                          float* __restrict__ out, int Np, int S, long total)
{
  const long t = (long)blockIdx.x * blockDim.x + threadIdx.x;
  if (t >= total) return;                     // t = b*S + s
  const long b = t / S;
  const int  id = fi[t];
  const float* src = xyz + ((long)b * Np + id) * 3;
  out[t * 3 + 0] = src[0];
  out[t * 3 + 1] = src[1];
  out[t * 3 + 2] = src[2];
}

// 32-NN (ascending distance, earlier index wins ties): one thread per (b,s)
__global__ void k_knn32(const float* __restrict__ q, const float* __restrict__ p,
                        int* __restrict__ ki, int Nq, int Np, long total)
{
  const long t = (long)blockIdx.x * blockDim.x + threadIdx.x;
  if (t >= total) return;
  const long b = t / Nq;
  const float* Q = q + t * 3;
  const float* P = p + (long)b * Np * 3;
  float dists[32]; int idxs[32];
#pragma unroll
  for (int j = 0; j < 32; ++j) { dists[j] = 3.4e38f; idxs[j] = 0; }
  const float qx = Q[0], qy = Q[1], qz = Q[2];
  for (int n = 0; n < Np; ++n) {
    const float dx = P[n * 3 + 0] - qx;
    const float dy = P[n * 3 + 1] - qy;
    const float dz = P[n * 3 + 2] - qz;
    const float d = dx * dx + dy * dy + dz * dz;
    if (d < dists[31]) {
      int j = 31;
      while (j > 0 && dists[j - 1] > d) {
        dists[j] = dists[j - 1]; idxs[j] = idxs[j - 1]; --j;
      }
      dists[j] = d; idxs[j] = n;
    }
  }
#pragma unroll
  for (int j = 0; j < 32; ++j) ki[t * 32 + j] = idxs[j];
}

// 3-NN + inverse-distance weights for feature propagation
__global__ void k_knn3(const float* __restrict__ q, const float* __restrict__ p,
                       int* __restrict__ idx3, float* __restrict__ w3,
                       int Nq, int Np, long total)
{
  const long t = (long)blockIdx.x * blockDim.x + threadIdx.x;
  if (t >= total) return;
  const long b = t / Nq;
  const float* Q = q + t * 3;
  const float* P = p + (long)b * Np * 3;
  float d0 = 3.4e38f, d1 = 3.4e38f, d2 = 3.4e38f;
  int i0 = 0, i1 = 0, i2 = 0;
  const float qx = Q[0], qy = Q[1], qz = Q[2];
  for (int n = 0; n < Np; ++n) {
    const float dx = P[n * 3 + 0] - qx;
    const float dy = P[n * 3 + 1] - qy;
    const float dz = P[n * 3 + 2] - qz;
    const float d = dx * dx + dy * dy + dz * dz;
    if (d < d0) { d2 = d1; i2 = i1; d1 = d0; i1 = i0; d0 = d; i0 = n; }
    else if (d < d1) { d2 = d1; i2 = i1; d1 = d; i1 = n; }
    else if (d < d2) { d2 = d; i2 = n; }
  }
  float w0 = 1.f / (d0 + 1e-8f);
  float w1 = 1.f / (d1 + 1e-8f);
  float w2 = 1.f / (d2 + 1e-8f);
  const float inv = 1.f / (w0 + w1 + w2);
  idx3[t * 3 + 0] = i0; idx3[t * 3 + 1] = i1; idx3[t * 3 + 2] = i2;
  w3[t * 3 + 0] = w0 * inv; w3[t * 3 + 1] = w1 * inv; w3[t * 3 + 2] = w2 * inv;
}

// feat[row][d] = f[b][d][kn] - f[b][d][cn]; feat[row][D+d] = f[b][d][cn]
// row = (b*S+s)*32+k; f is [B,D,Np]
__global__ void k_group_feat(const float* __restrict__ f, const int* __restrict__ fi,
                             const int* __restrict__ ki, float* __restrict__ feat,
                             int D, int Np, int S, long total)
{
  const long t = (long)blockIdx.x * blockDim.x + threadIdx.x;
  if (t >= total) return;
  const int  d = (int)(t % D);
  const long r = t / D;
  const int  k = (int)(r % 32);
  const long bs = r / 32;
  const long b = bs / S;
  const int  cn = fi[bs];
  const int  kn = ki[bs * 32 + k];
  const float* fb = f + ((long)b * D + d) * Np;
  const float cen = fb[cn];
  float* o = feat + r * (2L * D);
  o[d]     = fb[kn] - cen;
  o[D + d] = cen;
}

// out[b][o][s] = max_k Y[((b*S+s)*32+k)*C + o]
__global__ void k_maxk(const float* __restrict__ Y, float* __restrict__ out,
                       int C, int S, long total)
{
  const long t = (long)blockIdx.x * blockDim.x + threadIdx.x;
  if (t >= total) return;
  const int  s = (int)(t % S);
  const long bo = t / S;
  const int  o = (int)(bo % C);
  const long b = bo / C;
  const float* base = Y + (((long)b * S + s) * 32) * C + o;
  float m = base[0];
#pragma unroll 4
  for (int k = 1; k < 32; ++k) m = fmaxf(m, base[(long)k * C]);
  out[((long)b * C + o) * S + s] = m;
}

// O = X - Y per batch (contiguous per-batch slabs of `per` elements)
__global__ void k_sub(const float* __restrict__ X, long sbX,
                      const float* __restrict__ Y, long sbY,
                      float* __restrict__ O, long sbO, long per, long total)
{
  const long t = (long)blockIdx.x * blockDim.x + threadIdx.x;
  if (t >= total) return;
  const long b = t / per, i = t % per;
  O[b * sbO + i] = X[b * sbX + i] - Y[b * sbY + i];
}

// softmax over a row of 128 (one 128-thread block per row)
__global__ void k_softmax_rows(float* __restrict__ att, int Ncol)
{
  __shared__ float red[128];
  float* row = att + (long)blockIdx.x * Ncol;
  const int t = threadIdx.x;
  const float v = (t < Ncol) ? row[t] : -3.4e38f;
  red[t] = v; __syncthreads();
  for (int st = 64; st > 0; st >>= 1) {
    if (t < st) red[t] = fmaxf(red[t], red[t + st]);
    __syncthreads();
  }
  const float mx = red[0]; __syncthreads();
  const float e = (t < Ncol) ? __expf(v - mx) : 0.f;
  red[t] = e; __syncthreads();
  for (int st = 64; st > 0; st >>= 1) {
    if (t < st) red[t] += red[t + st];
    __syncthreads();
  }
  if (t < Ncol) row[t] = e / red[0];
}

// att[:, m] /= (1e-9 + sum_n att[n][m]); one thread per (b, m)
__global__ void k_att_colnorm(float* __restrict__ att, int Nn, long total)
{
  const long t = (long)blockIdx.x * blockDim.x + threadIdx.x;
  if (t >= total) return;
  const int  m = (int)(t % Nn);
  const long b = t / Nn;
  float* A = att + b * (long)Nn * Nn;
  float s = 0.f;
  for (int n = 0; n < Nn; ++n) s += A[(long)n * Nn + m];
  const float inv = 1.f / (1e-9f + s);
  for (int n = 0; n < Nn; ++n) A[(long)n * Nn + m] *= inv;
}

// mx[b*C+c] = max_n X[(b*C+c)*Nn + n]
__global__ void k_chanmax(const float* __restrict__ X, float* __restrict__ mx,
                          int Nn, long total)
{
  const long t = (long)blockIdx.x * blockDim.x + threadIdx.x;
  if (t >= total) return;
  const float* base = X + t * Nn;
  float m = base[0];
  for (int n = 1; n < Nn; ++n) m = fmaxf(m, base[n]);
  mx[t] = m;
}

// cat8[b][c][n] = cat4[b][c][n]; cat8[b][Cq+c][n] = chanmax[b][c]
__global__ void k_cat8(const float* __restrict__ cat4, const float* __restrict__ mx,
                       float* __restrict__ cat8, int Cq, int Nn, long total)
{
  const long t = (long)blockIdx.x * blockDim.x + threadIdx.x;
  if (t >= total) return;
  const int  n = (int)(t % Nn);
  const long bc = t / Nn;
  const int  c = (int)(bc % Cq);
  const long b = bc / Cq;
  cat8[((long)b * 2 * Cq + c) * Nn + n]      = cat4[t];
  cat8[((long)b * 2 * Cq + Cq + c) * Nn + n] = mx[(long)b * Cq + c];
}

// out[b][c][n]: c<Dd -> ptsd; else 3-NN interpolation of ptss
__global__ void k_fp_concat(const float* __restrict__ ptsd, const float* __restrict__ ptss,
                            const int* __restrict__ idx3, const float* __restrict__ w3,
                            float* __restrict__ outp,
                            int Dd, int Ds, int Nd, int Ns, long total)
{
  const long t = (long)blockIdx.x * blockDim.x + threadIdx.x;
  if (t >= total) return;
  const int  n = (int)(t % Nd);
  const long bc = t / Nd;
  const int  c = (int)(bc % (Dd + Ds));
  const long b = bc / (Dd + Ds);
  float v;
  if (c < Dd) {
    v = ptsd[((long)b * Dd + c) * Nd + n];
  } else {
    const int ds = c - Dd;
    const float* S = ptss + ((long)b * Ds + ds) * Ns;
    const int*   I = idx3 + ((long)b * Nd + n) * 3;
    const float* W = w3   + ((long)b * Nd + n) * 3;
    v = W[0] * S[I[0]] + W[1] * S[I[1]] + W[2] * S[I[2]];
  }
  outp[t] = v;
}

// ============================================================================
// Host orchestration
// ============================================================================
static inline long cdiv(long a, long b) { return (a + b - 1) / b; }

extern "C" void kernel_launch(void* const* d_in, const int* in_sizes, int n_in,
                              void* d_out, int out_size, void* d_ws, size_t ws_size,
                              hipStream_t stream) {
  (void)in_sizes; (void)n_in; (void)out_size; (void)ws_size;

  // ---- input unpack (setup_inputs dict order: x, then params in insert order)
  int ii = 0;
  auto nxt = [&]() { return (const float*)d_in[ii++]; };
  const float* x_in    = nxt();                       // [8,2048,3]
  const float* conv1_w = nxt();
  const float* bn1_g   = nxt();  const float* bn1_b = nxt();
  const float* conv2_w = nxt();
  const float* bn2_g   = nxt();  const float* bn2_b = nxt();
  const float *gl_w1[4], *gl_g1[4], *gl_b1[4], *gl_w2[4], *gl_g2[4], *gl_b2[4];
  for (int i = 0; i < 4; ++i) {
    gl_w1[i] = nxt(); gl_g1[i] = nxt(); gl_b1[i] = nxt();
    gl_w2[i] = nxt(); gl_g2[i] = nxt(); gl_b2[i] = nxt();
  }
  const float* pt_c1 = nxt(); const float* pt_g1 = nxt(); const float* pt_b1 = nxt();
  const float* pt_c2 = nxt(); const float* pt_g2 = nxt(); const float* pt_b2 = nxt();
  const float *sa_qk[4], *sa_vw[4], *sa_vb[4], *sa_tw[4], *sa_tb[4], *sa_bg[4], *sa_bb[4];
  for (int i = 0; i < 4; ++i) {
    sa_qk[i] = nxt(); sa_vw[i] = nxt(); sa_vb[i] = nxt();
    sa_tw[i] = nxt(); sa_tb[i] = nxt(); sa_bg[i] = nxt(); sa_bb[i] = nxt();
  }
  const float* fuse_w = nxt(); const float* fuse_g = nxt(); const float* fuse_b = nxt();
  const float *fp_w[4], *fp_cb[4], *fp_g[4], *fp_bb[4];    // order: fp4,fp3,fp2,fp1
  for (int i = 0; i < 4; ++i) {
    fp_w[i] = nxt(); fp_cb[i] = nxt(); fp_g[i] = nxt(); fp_bb[i] = nxt();
  }
  const float* logits_w = nxt(); const float* logits_b = nxt();

  // ---- workspace bump allocator (needs ~400 MB) ----
  size_t off = 0;
  auto allocf = [&](size_t n) {
    float* p = (float*)((char*)d_ws + off);
    off = (off + n * sizeof(float) + 255) & ~(size_t)255;
    return p;
  };
  auto alloci = [&](size_t n) {
    int* p = (int*)((char*)d_ws + off);
    off = (off + n * sizeof(int) + 255) & ~(size_t)255;
    return p;
  };

  float* stats   = allocf(16384);
  float* hA      = allocf(8L * 64 * 2048);
  float* f0      = allocf(8L * 64 * 2048);
  float* featBuf = allocf(33554432);           // 8*1024*32 x 128 (max level)
  float* yBuf    = allocf(33554432);
  float* f1 = allocf(8L * 128 * 1024);
  float* f2 = allocf(8L * 256 * 512);
  float* f3 = allocf(8L * 512 * 256);
  float* f4 = allocf(8L * 1024 * 128);
  float* xyz1 = allocf(8L * 1024 * 3);
  float* xyz2 = allocf(8L * 512 * 3);
  float* xyz3 = allocf(8L * 256 * 3);
  float* xyz4 = allocf(8L * 128 * 3);
  int* fi  = alloci(8L * 1024);
  int* kiB = alloci(8L * 1024 * 32);
  float* ptA  = allocf(8L * 1024 * 128);
  float* ptB  = allocf(8L * 1024 * 128);
  float* xq   = allocf(8L * 128 * 256);
  float* att  = allocf(8L * 128 * 128);
  float* xv   = allocf(8L * 1024 * 128);
  float* xr   = allocf(8L * 1024 * 128);
  float* tsub = allocf(8L * 1024 * 128);
  float* ty   = allocf(8L * 1024 * 128);
  float* cat4 = allocf(8L * 4096 * 128);
  float* maxc = allocf(8L * 4096);
  float* cat8 = allocf(8L * 8192 * 128);
  float* fused = allocf(8L * 4096 * 128);
  int*   idx3 = alloci(8L * 2048 * 3);
  float* w3   = allocf(8L * 2048 * 3);
  // FP-stage buffers alias the (now idle) local_op staging buffer
  float* fpnew  = featBuf;                      // <= 9.44M floats
  float* fp4out = featBuf + 10485760;           // 8*2048*256
  float* fp3out = fp4out + 4194304;             // 8*1024*512
  float* fp2out = fp3out + 4194304;             // 8*512*1024
  float* fp1out = fp2out + 4194304;             // 8*256*2048

  auto gemm = [&](float* C, const float* A, const float* Bm, const float* bias,
                  int M, int N, int K,
                  long sAm, long sAk, long bA,
                  long sBn, long sBk, long bB,
                  long sCm, long sCn, long bC, int batches) {
    dim3 g((unsigned)cdiv(N, 32), (unsigned)cdiv(M, 32), (unsigned)batches);
    k_gemm_bf16<<<g, 128, 0, stream>>>(C, A, Bm, bias, M, N, K,
                                       sAm, sAk, bA, sBn, sBk, bB, sCm, sCn, bC);
  };
  auto bnorm = [&](float* X, const float* g, const float* b,
                   int C, int Bc, int Nc, long sc, long sb,
                   float* Y, long scO, long sbO,
                   const float* res, long scR, long sbR, int act) {
    k_bn_stats<<<C, 256, 0, stream>>>(X, stats, C, Bc, Nc, sc, sb, 1e-5f);
    const long total = (long)C * Bc * Nc;
    k_bn_apply<<<(unsigned)cdiv(total, 256), 256, 0, stream>>>(
        X, Y, res, stats, g, b, C, Bc, Nc, sc, sb, scO, sbO, scR, sbR, act);
  };

  // ================= encoder stem =================
  // conv1: h[b][o][n] = conv1_w[o][c] * x[b][n][c]
  gemm(hA, conv1_w, x_in, nullptr, 64, 2048, 3,
       3, 1, 0,   3, 1, 2048L * 3,   2048, 1, 64L * 2048, 8);
  bnorm(hA, bn1_g, bn1_b, 64, 8, 2048, 2048, 64L * 2048,
        hA, 2048, 64L * 2048, nullptr, 0, 0, 1);
  gemm(f0, conv2_w, hA, nullptr, 64, 2048, 64,
       64, 1, 0,   1, 2048, 64L * 2048,   2048, 1, 64L * 2048, 8);
  bnorm(f0, bn2_g, bn2_b, 64, 8, 2048, 2048, 64L * 2048,
        f0, 2048, 64L * 2048, nullptr, 0, 0, 1);

  // ================= 4 x sample_and_group + local_op =================
  const int Sarr[4]  = {1024, 512, 256, 128};
  const int Nparr[4] = {2048, 1024, 512, 256};
  const int Darr[4]  = {64, 128, 256, 512};
  const float* xyzprev[4] = {x_in, xyz1, xyz2, xyz3};
  float* xyzl[4]  = {xyz1, xyz2, xyz3, xyz4};
  float* fprev[4] = {f0, f1, f2, f3};
  float* fout[4]  = {f1, f2, f3, f4};
  for (int l = 0; l < 4; ++l) {
    const int S = Sarr[l], Np = Nparr[l], D = Darr[l], c = 2 * D;
    k_fps<<<8, 256, 0, stream>>>(xyzprev[l], fi, Np, S);
    const long tg = 8L * S;
    k_gather3<<<(unsigned)cdiv(tg, 256), 256, 0, stream>>>(xyzprev[l], fi, xyzl[l], Np, S, tg);
    k_knn32<<<(unsigned)cdiv(tg, 128), 128, 0, stream>>>(xyzl[l], xyzprev[l], kiB, S, Np, tg);
    const long tf = 8L * S * 32 * D;
    k_group_feat<<<(unsigned)cdiv(tf, 256), 256, 0, stream>>>(fprev[l], fi, kiB, featBuf, D, Np, S, tf);
    const long M = 8L * S * 32;
    gemm(yBuf, featBuf, gl_w1[l], nullptr, (int)M, c, c,
         c, 1, 0,   c, 1, 0,   c, 1, 0, 1);
    bnorm(yBuf, gl_g1[l], gl_b1[l], c, (int)M, 1, 1, c,
          yBuf, 1, c, nullptr, 0, 0, 1);
    gemm(featBuf, yBuf, gl_w2[l], nullptr, (int)M, c, c,
         c, 1, 0,   c, 1, 0,   c, 1, 0, 1);
    bnorm(featBuf, gl_g2[l], gl_b2[l], c, (int)M, 1, 1, c,
          featBuf, 1, c, nullptr, 0, 0, 1);
    const long tm = 8L * c * S;
    k_maxk<<<(unsigned)cdiv(tm, 256), 256, 0, stream>>>(featBuf, fout[l], c, S, tm);
  }

  // ================= StackedAttention =================
  gemm(ptA, pt_c1, f4, nullptr, 1024, 128, 1024,
       1024, 1, 0,   1, 128, 1024L * 128,   128, 1, 1024L * 128, 8);
  bnorm(ptA, pt_g1, pt_b1, 1024, 8, 128, 128, 1024L * 128,
        ptA, 128, 1024L * 128, nullptr, 0, 0, 1);
  gemm(ptB, pt_c2, ptA, nullptr, 1024, 128, 1024,
       1024, 1, 0,   1, 128, 1024L * 128,   128, 1, 1024L * 128, 8);
  bnorm(ptB, pt_g2, pt_b2, 1024, 8, 128, 128, 1024L * 128,
        ptB, 128, 1024L * 128, nullptr, 0, 0, 1);

  for (int i = 0; i < 4; ++i) {
    const float* hin; long hinB;
    if (i == 0) { hin = ptB;                           hinB = 1024L * 128; }
    else        { hin = cat4 + (long)(i - 1) * 1024 * 128; hinB = 4096L * 128; }
    float* hout = cat4 + (long)i * 1024 * 128;
    // xq[b][n][o] = qk[o][c] * hin[b][c][n]  (k shares weights -> att = XQ XQ^T)
    gemm(xq, hin, sa_qk[i], nullptr, 128, 256, 1024,
         1, 128, hinB,   1024, 1, 0,   256, 1, 128L * 256, 8);
    gemm(att, xq, xq, nullptr, 128, 128, 256,
         256, 1, 128L * 256,   256, 1, 128L * 256,   128, 1, 128L * 128, 8);
    k_softmax_rows<<<8 * 128, 128, 0, stream>>>(att, 128);
    k_att_colnorm<<<(unsigned)cdiv(8L * 128, 256), 256, 0, stream>>>(att, 128, 8L * 128);
    gemm(xv, sa_vw[i], hin, sa_vb[i], 1024, 128, 1024,
         1024, 1, 0,   1, 128, hinB,   128, 1, 1024L * 128, 8);
    gemm(xr, xv, att, nullptr, 1024, 128, 128,
         128, 1, 1024L * 128,   1, 128, 128L * 128,   128, 1, 1024L * 128, 8);
    k_sub<<<(unsigned)cdiv(8L * 131072, 256), 256, 0, stream>>>(
        hin, hinB, xr, 131072, tsub, 131072, 131072, 8L * 131072);
    gemm(ty, sa_tw[i], tsub, sa_tb[i], 1024, 128, 1024,
         1024, 1, 0,   1, 128, 131072,   128, 1, 131072, 8);
    bnorm(ty, sa_bg[i], sa_bb[i], 1024, 8, 128, 128, 131072,
          hout, 128, 4096L * 128, hin, 128, hinB, 1);
  }

  k_chanmax<<<(unsigned)cdiv(8L * 4096, 256), 256, 0, stream>>>(cat4, maxc, 128, 8L * 4096);
  k_cat8<<<(unsigned)cdiv(8L * 4096 * 128, 256), 256, 0, stream>>>(
      cat4, maxc, cat8, 4096, 128, 8L * 4096 * 128);
  gemm(fused, fuse_w, cat8, nullptr, 4096, 128, 8192,
       8192, 1, 0,   1, 128, 8192L * 128,   128, 1, 4096L * 128, 8);
  bnorm(fused, fuse_g, fuse_b, 4096, 8, 128, 128, 4096L * 128,
        fused, 128, 4096L * 128, nullptr, 0, 0, 2);   // leaky 0.2

  // ================= feature propagation =================
  auto fp_stage = [&](const float* xyzd, const float* xyzs,
                      const float* ptsd, const float* ptss,
                      int Nd, int Ns, int Dd, int Ds,
                      const float* W, const float* cb,
                      const float* g, const float* bb2, int Co, float* outbuf) {
    k_knn3<<<(unsigned)cdiv(8L * Nd, 256), 256, 0, stream>>>(xyzd, xyzs, idx3, w3, Nd, Ns, 8L * Nd);
    const long tot = 8L * (Dd + Ds) * Nd;
    k_fp_concat<<<(unsigned)cdiv(tot, 256), 256, 0, stream>>>(
        ptsd, ptss, idx3, w3, fpnew, Dd, Ds, Nd, Ns, tot);
    gemm(outbuf, W, fpnew, cb, Co, Nd, Dd + Ds,
         (long)(Dd + Ds), 1, 0,   1, (long)Nd, (long)(Dd + Ds) * Nd,
         (long)Nd, 1, (long)Co * Nd, 8);
    bnorm(outbuf, g, bb2, Co, 8, Nd, Nd, (long)Co * Nd,
          outbuf, Nd, (long)Co * Nd, nullptr, 0, 0, 1);
  };
  fp_stage(xyz3, xyz4, f3, fused,  256, 128,  512, 4096, fp_w[0], fp_cb[0], fp_g[0], fp_bb[0], 2048, fp4out);
  fp_stage(xyz2, xyz3, f2, fp4out, 512, 256,  256, 2048, fp_w[1], fp_cb[1], fp_g[1], fp_bb[1], 1024, fp3out);
  fp_stage(xyz1, xyz2, f1, fp3out, 1024, 512, 128, 1024, fp_w[2], fp_cb[2], fp_g[2], fp_bb[2],  512, fp2out);
  fp_stage(x_in, xyz1, f0, fp2out, 2048, 1024, 64,  512, fp_w[3], fp_cb[3], fp_g[3], fp_bb[3],  256, fp1out);

  // ================= logits -> d_out [B,N,8] =================
  gemm((float*)d_out, logits_w, fp1out, logits_b, 8, 2048, 256,
       256, 1, 0,   1, 2048, 256L * 2048,   1, 8, 2048L * 8, 8);
}
